// MultiHeadAttention_8993661518300
// MI455X (gfx1250) — matrix operand order repaired and checked by reading, not compiled, not run on hardware
//
#include <hip/hip_runtime.h>
#include <hip/hip_bf16.h>

// B=4, N=2048, H=1024, NH=16, HD=64
typedef __attribute__((ext_vector_type(16))) _Float16 v16h;
typedef __attribute__((ext_vector_type(8)))  _Float16 v8h;
typedef __attribute__((ext_vector_type(4)))  _Float16 v4h;
typedef __attribute__((ext_vector_type(8)))  float    v8f;

#define USE_TDM (__has_builtin(__builtin_amdgcn_tensor_load_to_lds) && \
                 __has_builtin(__builtin_amdgcn_s_wait_tensorcnt))

__device__ __forceinline__ v8f zero_v8f() {
  v8f z;
#pragma unroll
  for (int i = 0; i < 8; ++i) z[i] = 0.0f;
  return z;
}

__device__ __forceinline__ v8f wmma16(v16h a, v16h b, v8f c) {
  // D = A(16x32 f16) * B(32x16 f16) + C(16x16 f32)
  return __builtin_amdgcn_wmma_f32_16x16x32_f16(
      false, a, false, b, (short)0, c, false, false);
}

// A-fragment (16x32, lane=row) from f16 row pointer; k0 includes +8*(lane>=16).
// One contiguous 16-half run at k0 + (k0 & 8) (= chunk base + 16*(lane>=16)).
__device__ __forceinline__ v16h load_a_f16(const _Float16* __restrict__ p, int k0) {
  v8h lo  = *(const v8h*)(p + k0 + (k0 & 8));
  v8h hi8 = *(const v8h*)(p + k0 + (k0 & 8) + 8);
  v16h a;
#pragma unroll
  for (int i = 0; i < 8; ++i) { a[i] = lo[i]; a[8 + i] = hi8[i]; }
  return a;
}

#if USE_TDM
typedef unsigned int u32x4 __attribute__((ext_vector_type(4)));
typedef int          i32x4 __attribute__((ext_vector_type(4)));
typedef int          i32x8 __attribute__((ext_vector_type(8)));

// Build a 2-D tensor D# (CDNA5 ISA) and issue
// tensor_load_to_lds: tile (tile1 rows x tile0 elems, 2-byte data) from a
// row-major tensor with row stride stride0, into LDS at lds_ptr (packed).
__device__ __forceinline__ void tdm_load_2d(void* lds_ptr, const void* gptr,
                                            unsigned td0, unsigned td1,
                                            unsigned tile0, unsigned tile1,
                                            unsigned stride0) {
  unsigned long long ga = (unsigned long long)(size_t)gptr;
  unsigned lds = (unsigned)(size_t)lds_ptr;   // low 32 bits = LDS byte offset
  u32x4 g0;
  g0.x = 1u;                                   // count=1 (valid), user mode
  g0.y = lds;                                  // lds_addr [63:32]
  g0.z = (unsigned)ga;                         // global_addr [95:64]
  g0.w = (unsigned)((ga >> 32) & 0x01ffffffu)  // global_addr [120:96]
         | (2u << 30);                         // type=2 ("image") [127:126]
  i32x8 g1;
  g1[0] = (int)(1u << 16);                                  // data_size=2B
  g1[1] = (int)((td0 & 0xffffu) << 16);                     // tensor_dim0 lo16
  g1[2] = (int)(((td0 >> 16) & 0xffffu) | ((td1 & 0xffffu) << 16));
  g1[3] = (int)(((td1 >> 16) & 0xffffu) | ((tile0 & 0xffffu) << 16));
  g1[4] = (int)(tile1 & 0xffffu);                           // tile_dim1; dim2=0
  g1[5] = (int)stride0;                                     // dim0_stride lo32
  g1[6] = 0;
  g1[7] = 0;
  i32x4 z4 = {0, 0, 0, 0};
#if __clang_major__ >= 23
  i32x8 z8 = {0, 0, 0, 0, 0, 0, 0, 0};
  __builtin_amdgcn_tensor_load_to_lds(g0, g1, z4, z4, z8, 0);
#else
  __builtin_amdgcn_tensor_load_to_lds(g0, g1, z4, z4, 0);
#endif
}
#endif  // USE_TDM

// -------------------- Kernel 0: f32 -> f16 conversion --------------------
__global__ __launch_bounds__(256) void
cvt_f16_kernel(const float* __restrict__ in, _Float16* __restrict__ out, int n4) {
  const int i = blockIdx.x * 256 + threadIdx.x;
  if (i < n4) {
    float4 f = ((const float4*)in)[i];
    v4h o;
    o[0] = (_Float16)f.x; o[1] = (_Float16)f.y;
    o[2] = (_Float16)f.z; o[3] = (_Float16)f.w;
    ((v4h*)out)[i] = o;
  }
}

// -------------------- Kernel 1: QKV projection --------------------
// y[8192,3072] = x_f16[8192,1024] @ w_f16[3072,1024]^T + b_qkv
// Scatter to f16 Q/K [bh,n,64] and V^T [bh,64,n]. One wave = 16x64 tile,
// register double-buffered over the K dimension.
__global__ __launch_bounds__(128) void
qkv_proj_kernel(const _Float16* __restrict__ xh, const _Float16* __restrict__ wh,
                const float* __restrict__ bias,
                _Float16* __restrict__ qbuf, _Float16* __restrict__ kbuf,
                _Float16* __restrict__ vbuf)
{
  const int lane = threadIdx.x & 31;
  const int l16  = lane & 15;
  const int hi   = lane >> 4;
  const int gw   = blockIdx.x * 4 + (threadIdx.x >> 5);
  const int rt   = gw & 511;        // 512 row tiles (8192/16)
  const int ct   = gw >> 9;         // 48 col tiles (3072/64)
  const int row0 = rt * 16;
  const int col0 = ct * 64;

  const _Float16* xrow = xh + (size_t)(row0 + l16) * 1024;
  const _Float16* wr0  = wh + (size_t)(col0 + l16) * 1024 + 16 * hi;

  v8f acc[4];
#pragma unroll
  for (int c = 0; c < 4; ++c) acc[c] = zero_v8f();

  v16h a_c  = load_a_f16(xrow, 8 * hi);
  v16h b_c0 = *(const v16h*)(wr0);
  v16h b_c1 = *(const v16h*)(wr0 + 16384);
  v16h b_c2 = *(const v16h*)(wr0 + 32768);
  v16h b_c3 = *(const v16h*)(wr0 + 49152);

  for (int kc = 0; kc < 1024 - 32; kc += 32) {
    v16h a_n  = load_a_f16(xrow, kc + 32 + 8 * hi);
    v16h b_n0 = *(const v16h*)(wr0 + kc + 32);
    v16h b_n1 = *(const v16h*)(wr0 + 16384 + kc + 32);
    v16h b_n2 = *(const v16h*)(wr0 + 32768 + kc + 32);
    v16h b_n3 = *(const v16h*)(wr0 + 49152 + kc + 32);
    acc[0] = wmma16(a_c, b_c0, acc[0]);
    acc[1] = wmma16(a_c, b_c1, acc[1]);
    acc[2] = wmma16(a_c, b_c2, acc[2]);
    acc[3] = wmma16(a_c, b_c3, acc[3]);
    a_c = a_n; b_c0 = b_n0; b_c1 = b_n1; b_c2 = b_n2; b_c3 = b_n3;
  }
  acc[0] = wmma16(a_c, b_c0, acc[0]);
  acc[1] = wmma16(a_c, b_c1, acc[1]);
  acc[2] = wmma16(a_c, b_c2, acc[2]);
  acc[3] = wmma16(a_c, b_c3, acc[3]);

#pragma unroll
  for (int c = 0; c < 4; ++c) {
    const int oc    = col0 + c * 16 + l16;
    const float bv  = bias[oc];
    const int which = oc >> 10;        // 0=Q 1=K 2=V
    const int hl    = oc & 1023;
    const int head  = hl >> 6;
    const int d     = hl & 63;
#pragma unroll
    for (int r = 0; r < 8; ++r) {
      const int tok = row0 + r + 8 * hi;
      const int bb  = tok >> 11;
      const int n   = tok & 2047;
      const _Float16 hv = (_Float16)(acc[c][r] + bv);
      const size_t bh = (size_t)(bb * 16 + head);
      if (which == 0)      qbuf[(bh * 2048 + n) * 64 + d] = hv;
      else if (which == 1) kbuf[(bh * 2048 + n) * 64 + d] = hv;
      else                 vbuf[(bh * 64 + d) * 2048 + n] = hv;   // transposed
    }
  }
}

// -------------------- Kernel 2: flash attention --------------------
// One wave per (b,h,32-query tile). K/V blocks staged to LDS by the Tensor
// Data Mover (double-buffered, TENSORcnt); online softmax over 64 blocks of
// 32 keys. 16 WMMAs per block; softmax row pass uses all 32 lanes (row=lane).
__global__ __launch_bounds__(32) void
flash_attn_kernel(const _Float16* __restrict__ qbuf, const _Float16* __restrict__ kbuf,
                  const _Float16* __restrict__ vbuf, _Float16* __restrict__ abuf)
{
  const int lane = threadIdx.x;
  const int l16  = lane & 15;
  const int hi   = lane >> 4;
  const int qt   = blockIdx.x & 63;    // 64 tiles of 32 queries (2048/32)
  const int bh   = blockIdx.x >> 6;    // b*16+h
  const int bb   = bh >> 4;
  const int h    = bh & 15;

  const _Float16* Q = qbuf + ((size_t)bh * 2048 + (size_t)qt * 32) * 64;
  const _Float16* K = kbuf + (size_t)bh * (2048 * 64);
  const _Float16* V = vbuf + (size_t)bh * (64 * 2048);   // V^T: row=d, col=key

  __shared__ float sS[32 * 33];                   // padded, conflict-free rows
  __shared__ __align__(16) _Float16 sP[32 * 32];  // probabilities, f16
  __shared__ float sStat[64];                     // [0..31]=corr, [32..63]=1/l
#if USE_TDM
  __shared__ __align__(16) _Float16 sK[2][32 * 64];   // 32 keys x 64 dims
  __shared__ __align__(16) _Float16 sV[2][64 * 32];   // 64 dims x 32 keys
#endif

  // Q fragments: 2 q-row-tiles x 2 k-chunks; fold softmax scale 2^-3 (exact).
  v16h qf[2][2];
#pragma unroll
  for (int qI = 0; qI < 2; ++qI) {
#pragma unroll
    for (int c = 0; c < 2; ++c) {
      v16h t = load_a_f16(Q + (size_t)(qI * 16 + l16) * 64, c * 32 + 8 * hi);
#pragma unroll
      for (int i = 0; i < 16; ++i) t[i] = t[i] * (_Float16)0.125f;
      qf[qI][c] = t;
    }
  }

  v8f o[2][4];
#pragma unroll
  for (int qI = 0; qI < 2; ++qI)
#pragma unroll
    for (int c = 0; c < 4; ++c) o[qI][c] = zero_v8f();
  float m_state = -3.0e38f;   // row = lane (0..31)
  float l_state = 0.0f;

#if USE_TDM
  tdm_load_2d(&sK[0][0], K, 64,   2048, 64, 32, 64);
  tdm_load_2d(&sV[0][0], V, 2048, 64,   32, 64, 2048);
  int buf = 0;
#endif

  for (int kb = 0; kb < 2048; kb += 32) {
#if USE_TDM
    if (kb + 32 < 2048) {
      tdm_load_2d(&sK[buf ^ 1][0], K + (size_t)(kb + 32) * 64, 64, 2048, 64, 32, 64);
      tdm_load_2d(&sV[buf ^ 1][0], V + (kb + 32),              2048, 64, 32, 64, 2048);
      __builtin_amdgcn_s_wait_tensorcnt((short)2);   // current buffer complete
    } else {
      __builtin_amdgcn_s_wait_tensorcnt((short)0);
    }
    asm volatile("" ::: "memory");
    const _Float16* kbase = &sK[buf][0];   // stride 64
    const _Float16* vbase = &sV[buf][0];
    const int vstride = 32;
#else
    const _Float16* kbase = K + (size_t)kb * 64;
    const _Float16* vbase = V + kb;
    const int vstride = 2048;
#endif
    // S(32x32) = Q(32x64) K_block^T(64x32): 2 q-tiles x 2 key-tiles
    v8f s[2][2];
#pragma unroll
    for (int qI = 0; qI < 2; ++qI) { s[qI][0] = zero_v8f(); s[qI][1] = zero_v8f(); }
    {
      const _Float16* k0row = kbase + (size_t)l16 * 64;
      const _Float16* k1row = k0row + 16 * 64;
#pragma unroll
      for (int c = 0; c < 2; ++c) {
        v16h b0 = *(const v16h*)(k0row + c * 32 + 16 * hi);
        v16h b1 = *(const v16h*)(k1row + c * 32 + 16 * hi);
        s[0][0] = wmma16(qf[0][c], b0, s[0][0]);
        s[0][1] = wmma16(qf[0][c], b1, s[0][1]);
        s[1][0] = wmma16(qf[1][c], b0, s[1][0]);
        s[1][1] = wmma16(qf[1][c], b1, s[1][1]);
      }
    }
#pragma unroll
    for (int qI = 0; qI < 2; ++qI)
#pragma unroll
      for (int r = 0; r < 8; ++r) {
        sS[(qI * 16 + r + 8 * hi) * 33 + l16]      = s[qI][0][r];
        sS[(qI * 16 + r + 8 * hi) * 33 + 16 + l16] = s[qI][1][r];
      }
    __syncthreads();
    // row pass, full EXEC: lane i owns row i (32 rows)
    {
      const float* srow = sS + lane * 33;
      float mx = m_state;
#pragma unroll
      for (int j = 0; j < 32; ++j) mx = fmaxf(mx, srow[j]);
      const float corr = __expf(m_state - mx);
      float sum = 0.0f;
      _Float16* prow = sP + lane * 32;
#pragma unroll
      for (int j = 0; j < 32; ++j) {
        const float p = __expf(srow[j] - mx);
        sum += p;
        prow[j] = (_Float16)p;
      }
      l_state = l_state * corr + sum;
      m_state = mx;
      sStat[lane] = corr;
    }
    __syncthreads();
#pragma unroll
    for (int qI = 0; qI < 2; ++qI)
#pragma unroll
      for (int r = 0; r < 8; ++r) {
        const float cr = sStat[qI * 16 + r + 8 * hi];
        o[qI][0][r] *= cr; o[qI][1][r] *= cr;
        o[qI][2][r] *= cr; o[qI][3][r] *= cr;
      }
    v16h pf0 = load_a_f16(sP + (size_t)l16 * 32, 8 * hi);
    v16h pf1 = load_a_f16(sP + (size_t)(16 + l16) * 32, 8 * hi);
    // O(32x64) += P(32x32) V_block(32x64); V fragment shared by both q-tiles
#pragma unroll
    for (int c = 0; c < 4; ++c) {
      v16h vf = *(const v16h*)(vbase + (size_t)(c * 16 + l16) * vstride + 16 * hi);
      o[0][c] = wmma16(pf0, vf, o[0][c]);
      o[1][c] = wmma16(pf1, vf, o[1][c]);
    }
    __syncthreads();
#if USE_TDM
    buf ^= 1;
#endif
  }

  sStat[32 + lane] = 1.0f / l_state;
  __syncthreads();

  _Float16* orow = abuf + ((size_t)(bb * 2048 + qt * 32)) * 1024 + h * 64;
#pragma unroll
  for (int qI = 0; qI < 2; ++qI)
#pragma unroll
    for (int r = 0; r < 8; ++r) {
      const int m = qI * 16 + r + 8 * hi;
      const float li = sStat[32 + m];
#pragma unroll
      for (int c = 0; c < 4; ++c) {
        orow[(size_t)m * 1024 + c * 16 + l16] = (_Float16)(o[qI][c][r] * li);
      }
    }
}

// -------------------- Kernel 3: output projection --------------------
// out[8192,1024] = attn_f16[8192,1024] @ w_out_f16[1024,1024]^T + b_out (f32)
__global__ __launch_bounds__(128) void
out_proj_kernel(const _Float16* __restrict__ attn, const _Float16* __restrict__ wh,
                const float* __restrict__ bias, float* __restrict__ out)
{
  const int lane = threadIdx.x & 31;
  const int l16  = lane & 15;
  const int hi   = lane >> 4;
  const int gw   = blockIdx.x * 4 + (threadIdx.x >> 5);
  const int rt   = gw & 511;        // 512 row tiles
  const int ct   = gw >> 9;         // 16 col tiles (1024/64)
  const int row0 = rt * 16;
  const int col0 = ct * 64;

  const _Float16* arow = attn + (size_t)(row0 + l16) * 1024;
  const _Float16* wr0  = wh + (size_t)(col0 + l16) * 1024 + 16 * hi;

  v8f acc[4];
#pragma unroll
  for (int c = 0; c < 4; ++c) acc[c] = zero_v8f();

  v16h a_c  = load_a_f16(arow, 8 * hi);
  v16h b_c0 = *(const v16h*)(wr0);
  v16h b_c1 = *(const v16h*)(wr0 + 16384);
  v16h b_c2 = *(const v16h*)(wr0 + 32768);
  v16h b_c3 = *(const v16h*)(wr0 + 49152);

  for (int kc = 0; kc < 1024 - 32; kc += 32) {
    v16h a_n  = load_a_f16(arow, kc + 32 + 8 * hi);
    v16h b_n0 = *(const v16h*)(wr0 + kc + 32);
    v16h b_n1 = *(const v16h*)(wr0 + 16384 + kc + 32);
    v16h b_n2 = *(const v16h*)(wr0 + 32768 + kc + 32);
    v16h b_n3 = *(const v16h*)(wr0 + 49152 + kc + 32);
    acc[0] = wmma16(a_c, b_c0, acc[0]);
    acc[1] = wmma16(a_c, b_c1, acc[1]);
    acc[2] = wmma16(a_c, b_c2, acc[2]);
    acc[3] = wmma16(a_c, b_c3, acc[3]);
    a_c = a_n; b_c0 = b_n0; b_c1 = b_n1; b_c2 = b_n2; b_c3 = b_n3;
  }
  acc[0] = wmma16(a_c, b_c0, acc[0]);
  acc[1] = wmma16(a_c, b_c1, acc[1]);
  acc[2] = wmma16(a_c, b_c2, acc[2]);
  acc[3] = wmma16(a_c, b_c3, acc[3]);

#pragma unroll
  for (int c = 0; c < 4; ++c) {
    const int oc   = col0 + c * 16 + l16;
    const float bv = bias[oc];
#pragma unroll
    for (int r = 0; r < 8; ++r) {
      out[(size_t)(row0 + r + 8 * hi) * 1024 + oc] = acc[c][r] + bv;
    }
  }
}

extern "C" void kernel_launch(void* const* d_in, const int* in_sizes, int n_in,
                              void* d_out, int out_size, void* d_ws, size_t ws_size,
                              hipStream_t stream) {
  const float* x     = (const float*)d_in[0];  // [4,2048,1024]
  const float* w_qkv = (const float*)d_in[1];  // [3072,1024]
  const float* b_qkv = (const float*)d_in[2];  // [3072]
  const float* w_out = (const float*)d_in[3];  // [1024,1024]
  const float* b_out = (const float*)d_in[4];  // [1024]
  float* out = (float*)d_out;                  // [4,2048,1024] f32

  const size_t NX   = (size_t)8192 * 1024;     // 8,388,608
  const size_t NWQ  = (size_t)3072 * 1024;     // 3,145,728
  const size_t NWO  = (size_t)1024 * 1024;     // 1,048,576
  _Float16* p     = (_Float16*)d_ws;
  _Float16* xh    = p; p += NX;
  _Float16* wqkvh = p; p += NWQ;
  _Float16* wouth = p; p += NWO;
  _Float16* qbuf  = p; p += NX;
  _Float16* kbuf  = p; p += NX;
  _Float16* vbuf  = p; p += NX;                // transposed V
  _Float16* abuf  = p; p += NX;                // attention out (f16)

  // one-time f32->f16 conversions (bandwidth-bound, ~30 MB total)
  cvt_f16_kernel<<<dim3((unsigned)(NX / 4 / 256)),  dim3(256), 0, stream>>>(x,     xh,    (int)(NX / 4));
  cvt_f16_kernel<<<dim3((unsigned)(NWQ / 4 / 256)), dim3(256), 0, stream>>>(w_qkv, wqkvh, (int)(NWQ / 4));
  cvt_f16_kernel<<<dim3((unsigned)(NWO / 4 / 256)), dim3(256), 0, stream>>>(w_out, wouth, (int)(NWO / 4));

  // 512*48 = 24576 waves, 4 waves/block
  qkv_proj_kernel<<<dim3(6144), dim3(128), 0, stream>>>(xh, wqkvh, b_qkv, qbuf, kbuf, vbuf);
  // 4*16*64 = 4096 (b,h,32-query-tile) waves, 1 wave/block
  flash_attn_kernel<<<dim3(4096), dim3(32), 0, stream>>>(qbuf, kbuf, vbuf, abuf);
  // 512*16 = 8192 waves, 4 waves/block
  out_proj_kernel<<<dim3(2048), dim3(128), 0, stream>>>(abuf, wouth, b_out, out);
}
